// SelfAttention_53283364274429
// MI455X (gfx1250) — matrix operand order repaired and checked by reading, not compiled, not run on hardware
//
#include <hip/hip_runtime.h>

typedef __bf16 bf16_t;
typedef __attribute__((ext_vector_type(16))) __bf16 v16bf;
typedef __attribute__((ext_vector_type(8)))  __bf16 v8bf;
typedef __attribute__((ext_vector_type(4)))  __bf16 v4bf;
typedef __attribute__((ext_vector_type(8)))  float  v8f;

// Types matching the async-to-LDS builtin signature: AS(1)/AS(3) int4*.
typedef int v4i_t __attribute__((vector_size(16)));
typedef __attribute__((address_space(1))) v4i_t gv4i_t;
typedef __attribute__((address_space(3))) v4i_t lv4i_t;

#define EMBED    1024
#define HEADS    16
#define HEAD_DIM 64
#define SEQ      2048
#define BATCH    4

#if defined(__has_builtin)
#if __has_builtin(__builtin_amdgcn_global_load_async_to_lds_b128)
#define HAVE_ASYNC_LDS 1
#endif
#endif

__device__ __forceinline__ bf16_t f2bf(float x) { return (bf16_t)x; }

// Combine two contiguous 8-element LDS reads into one 16-element fragment.
__device__ __forceinline__ v16bf frag16(const bf16_t* p0, const bf16_t* p1) {
  v8bf lo = *(const v8bf*)p0;
  v8bf hi = *(const v8bf*)p1;
  return __builtin_shufflevector(lo, hi, 0, 1, 2, 3, 4, 5, 6, 7,
                                 8, 9, 10, 11, 12, 13, 14, 15);
}

__device__ __forceinline__ v8f wmma_bf16(v16bf a, v16bf b, v8f c) {
  return __builtin_amdgcn_wmma_f32_16x16x32_bf16(false, a, false, b, (short)0, c,
                                                 false, false);
}

// 16-byte global -> LDS copy: async (ASYNCcnt) when available, else via VGPRs.
__device__ __forceinline__ void copy16_g2l(const bf16_t* g, bf16_t* l) {
#if defined(HAVE_ASYNC_LDS)
  __builtin_amdgcn_global_load_async_to_lds_b128((gv4i_t*)g, (lv4i_t*)l, 0, 0);
#else
  *(v8bf*)l = *(const v8bf*)g;
#endif
}

__device__ __forceinline__ void wait_async0() {
#if defined(HAVE_ASYNC_LDS)
#if __has_builtin(__builtin_amdgcn_s_wait_asynccnt)
  __builtin_amdgcn_s_wait_asynccnt(0);
#else
  asm volatile("s_wait_asynccnt 0x0" ::: "memory");
#endif
#endif
}

// ---------------------------------------------------------------------------
// Elementwise f32 -> bf16 (vectorized: float4 in, 4x bf16 out)
// ---------------------------------------------------------------------------
__global__ __launch_bounds__(256) void cvt_f32_to_bf16(
    const float* __restrict__ x, bf16_t* __restrict__ y, int n4) {
  int i = blockIdx.x * 256 + threadIdx.x;
  if (i < n4) {
    const float4 v = ((const float4*)x)[i];
    v4bf o = {(bf16_t)v.x, (bf16_t)v.y, (bf16_t)v.z, (bf16_t)v.w};
    ((v4bf*)y)[i] = o;
  }
}

// ---------------------------------------------------------------------------
// GEMM: Y[M,N] = X[M,K](bf16) * W[K,N](bf16) + bias(f32)
// Block tile 128x128, BK=32, 256 threads = 8 waves in 4(M) x 2(N) grid.
// Double-buffered LDS: async-stage tile t+1 while WMMAs consume tile t.
// B staged TRANSPOSED so each B fragment is two contiguous ds_load_b128.
// ---------------------------------------------------------------------------
template <bool OUT_BF16>
__global__ __launch_bounds__(256) void gemm_bias_wmma(
    const bf16_t* __restrict__ X, const bf16_t* __restrict__ W,
    const float* __restrict__ bias, void* __restrict__ Yv,
    int M, int N, int K) {
  constexpr int BM = 128, BN = 128, BK = 32, PAD = 8;
  __shared__ bf16_t As[2][BM][BK + PAD];    // row = 80B (16B-aligned stride)
  __shared__ bf16_t BsT[2][BN][BK + PAD];   // transposed: [n][k], row = 80B

  const int tid  = threadIdx.x;
  const int wid  = tid >> 5;
  const int lane = tid & 31;
  const int hf   = lane >> 4;
  const int idx  = lane & 15;
  const int wr   = (wid & 3) * 32;      // wave row offset in block tile
  const int wc   = (wid >> 2) * 64;     // wave col offset in block tile

  const int n0 = blockIdx.x * BN;
  const int m0 = blockIdx.y * BM;

  auto stage = [&](int buf, int k0) {
#pragma unroll
    for (int i = tid; i < (BM * BK) / 8; i += 256) {   // A: async b128
      int r = i >> 2, c = (i & 3) * 8;
      copy16_g2l(&X[(size_t)(m0 + r) * K + k0 + c], &As[buf][r][c]);
    }
#pragma unroll
    for (int i = tid; i < (BK * BN) / 8; i += 256) {   // B: transposed scatter
      int r = i >> 4, c = (i & 15) * 8;
      v8bf w8 = *(const v8bf*)&W[(size_t)(k0 + r) * N + n0 + c];
#pragma unroll
      for (int j = 0; j < 8; ++j) BsT[buf][c + j][r] = w8[j];
    }
  };

  v8f acc[2][4] = {};
  const int KT = K / BK;

  stage(0, 0);
  wait_async0();
  __syncthreads();

  for (int kt = 0; kt < KT; ++kt) {
    const int cur = kt & 1, nxt = cur ^ 1;
    if (kt + 1 < KT) {
      stage(nxt, (kt + 1) * BK);
      if (kt + 2 < KT) {  // keep L2 two tiles ahead of the async engine
        int kp = (kt + 2) * BK;
        __builtin_prefetch(&X[(size_t)(m0 + (tid >> 1)) * K + kp + (tid & 1) * 16], 0, 3);
        __builtin_prefetch(&W[(size_t)(kp + (tid >> 4)) * N + n0 + (tid & 15) * 8], 0, 3);
      }
    }

    // A fragments: two contiguous 16B runs at k = hf*8 and 16+hf*8
    v16bf a0 = frag16(&As[cur][wr + idx][hf * 8], &As[cur][wr + idx][16 + hf * 8]);
    v16bf a1 = frag16(&As[cur][wr + 16 + idx][hf * 8],
                      &As[cur][wr + 16 + idx][16 + hf * 8]);

#pragma unroll
    for (int t = 0; t < 4; ++t) {
      const bf16_t* bp = &BsT[cur][wc + t * 16 + idx][hf * 8];
      v16bf b = frag16(bp, bp + 16);
      acc[0][t] = wmma_bf16(a0, b, acc[0][t]);
      acc[1][t] = wmma_bf16(a1, b, acc[1][t]);
    }

    wait_async0();
    __syncthreads();
  }

#pragma unroll
  for (int u = 0; u < 2; ++u) {
#pragma unroll
    for (int t = 0; t < 4; ++t) {
      int n = n0 + wc + t * 16 + idx;
      float bb = bias[n];
#pragma unroll
      for (int r = 0; r < 8; ++r) {
        int m = m0 + wr + u * 16 + r + hf * 8;
        float y = acc[u][t][r] + bb;
        if constexpr (OUT_BF16)
          ((bf16_t*)Yv)[(size_t)m * N + n] = f2bf(y);
        else
          ((float*)Yv)[(size_t)m * N + n] = y;
      }
    }
  }
}

// ---------------------------------------------------------------------------
// Flash attention: block = (batch n, head h, 128-query tile), 8 waves.
// Double-buffered K/V tiles; Q/K async-staged; V transposed for b128 frags.
// ---------------------------------------------------------------------------
__global__ __launch_bounds__(256) void flash_attn_wmma(
    const bf16_t* __restrict__ Q, const bf16_t* __restrict__ Kb,
    const bf16_t* __restrict__ V, const int* __restrict__ mask,
    bf16_t* __restrict__ O) {
  constexpr int BQ = 128, BKV = 32, D = HEAD_DIM, PAD = 8;
  __shared__ bf16_t Qs[BQ][D + PAD];         // row = 144B
  __shared__ bf16_t Ks[2][BKV][D + PAD];
  __shared__ bf16_t VsT[2][D][BKV + PAD];    // transposed: [d][key], row = 80B
  __shared__ bf16_t Ps[8][16][BKV + PAD];    // per-wave P re-layout buffer

  const int tid  = threadIdx.x;
  const int wid  = tid >> 5;
  const int lane = tid & 31;
  const int hf   = lane >> 4;
  const int idx  = lane & 15;

  const int q0 = blockIdx.x * BQ;
  const int h  = blockIdx.y;
  const int n  = blockIdx.z;

  const size_t base = (size_t)n * SEQ * EMBED + (size_t)h * D;

  auto stageKV = [&](int buf, int kv) {
#pragma unroll
    for (int i = tid; i < (BKV * D) / 8; i += 256) {
      int r = i >> 3, c = (i & 7) * 8;
      copy16_g2l(&Kb[base + (size_t)(kv + r) * EMBED + c], &Ks[buf][r][c]);
      v8bf v8 = *(const v8bf*)&V[base + (size_t)(kv + r) * EMBED + c];
#pragma unroll
      for (int j = 0; j < 8; ++j) VsT[buf][c + j][r] = v8[j];
    }
  };

  // stage Q tile + first K/V tile under one wait
#pragma unroll
  for (int i = tid; i < (BQ * D) / 8; i += 256) {
    int r = i >> 3, c = (i & 7) * 8;
    copy16_g2l(&Q[base + (size_t)(q0 + r) * EMBED + c], &Qs[r][c]);
  }
  stageKV(0, 0);
  wait_async0();
  __syncthreads();

  v16bf aQ[2];
#pragma unroll
  for (int kk = 0; kk < 2; ++kk)
    aQ[kk] = frag16(&Qs[wid * 16 + idx][kk * 32 + hf * 8],
                    &Qs[wid * 16 + idx][kk * 32 + 16 + hf * 8]);

  v8f acc[4] = {};
  float mrow[8], lrow[8];
#pragma unroll
  for (int r = 0; r < 8; ++r) { mrow[r] = -3.0e30f; lrow[r] = 0.0f; }

  // softmax over energy/sqrt(EMBED); work in exp2 domain
  const float sc = 0.03125f * 1.44269504088896340736f;

  const int NKV = SEQ / BKV;
  for (int kt = 0; kt < NKV; ++kt) {
    const int cur = kt & 1, nxt = cur ^ 1;
    const int kv = kt * BKV;
    if (kt + 1 < NKV) {
      stageKV(nxt, kv + BKV);
      if (kt + 2 < NKV) {
        size_t pref = base + (size_t)(kv + 2 * BKV + (tid >> 3)) * EMBED + (tid & 7) * 8;
        __builtin_prefetch(&Kb[pref], 0, 3);
        __builtin_prefetch(&V[pref], 0, 3);
      }
    }

    // S = Q * K^T : two 16x16 key tiles, d reduced in two K=32 steps
    v8f s0 = {}, s1 = {};
#pragma unroll
    for (int kk = 0; kk < 2; ++kk) {
      const bf16_t* k0p = &Ks[cur][idx][kk * 32 + hf * 8];
      const bf16_t* k1p = &Ks[cur][16 + idx][kk * 32 + hf * 8];
      v16bf b0 = frag16(k0p, k0p + 16);
      v16bf b1 = frag16(k1p, k1p + 16);
      s0 = wmma_bf16(aQ[kk], b0, s0);
      s1 = wmma_bf16(aQ[kk], b1, s1);
    }

    const int mk0 = mask[(size_t)n * SEQ + kv + idx];
    const int mk1 = mask[(size_t)n * SEQ + kv + 16 + idx];

#pragma unroll
    for (int r = 0; r < 8; ++r) {
      float z0 = mk0 ? s0[r] * sc : -3.0e30f;
      float z1 = mk1 ? s1[r] * sc : -3.0e30f;
      float zm = fmaxf(z0, z1);
#pragma unroll
      for (int off = 1; off <= 8; off <<= 1)
        zm = fmaxf(zm, __shfl_xor(zm, off, 32));   // stays within 16-lane half
      float mn    = fmaxf(mrow[r], zm);
      float alpha = exp2f(mrow[r] - mn);
      float p0    = exp2f(z0 - mn);
      float p1    = exp2f(z1 - mn);
      float ps    = p0 + p1;
#pragma unroll
      for (int off = 1; off <= 8; off <<= 1)
        ps += __shfl_xor(ps, off, 32);
      lrow[r] = lrow[r] * alpha + ps;
      mrow[r] = mn;
#pragma unroll
      for (int t = 0; t < 4; ++t) acc[t][r] *= alpha;
      int m = r + hf * 8;                          // C-layout row -> P row
      Ps[wid][m][idx]      = f2bf(p0);
      Ps[wid][m][16 + idx] = f2bf(p1);
    }
    __builtin_amdgcn_wave_barrier();   // per-wave LDS: hw in-order, fence compiler

    v16bf pa = frag16(&Ps[wid][idx][hf * 8], &Ps[wid][idx][16 + hf * 8]);
    __builtin_amdgcn_wave_barrier();

    // O += P * V : 4 output d-tiles, K=32 over this key block
#pragma unroll
    for (int t = 0; t < 4; ++t) {
      const bf16_t* vp = &VsT[cur][t * 16 + idx][hf * 8];
      v16bf bv = frag16(vp, vp + 16);
      acc[t] = wmma_bf16(pa, bv, acc[t]);
    }

    wait_async0();
    __syncthreads();
  }

#pragma unroll
  for (int t = 0; t < 4; ++t) {
#pragma unroll
    for (int r = 0; r < 8; ++r) {
      int m = q0 + wid * 16 + r + hf * 8;
      float inv = 1.0f / lrow[r];
      O[base + (size_t)m * EMBED + t * 16 + idx] = f2bf(acc[t][r] * inv);
    }
  }
}

// ---------------------------------------------------------------------------
extern "C" void kernel_launch(void* const* d_in, const int* in_sizes, int n_in,
                              void* d_out, int out_size, void* d_ws, size_t ws_size,
                              hipStream_t stream) {
  (void)in_sizes; (void)n_in; (void)out_size; (void)ws_size;

  const float* values = (const float*)d_in[0];
  const float* keys   = (const float*)d_in[1];
  const float* query  = (const float*)d_in[2];
  const int*   mask   = (const int*)d_in[3];
  const float* Wv = (const float*)d_in[4];  const float* bv = (const float*)d_in[5];
  const float* Wk = (const float*)d_in[6];  const float* bk = (const float*)d_in[7];
  const float* Wq = (const float*)d_in[8];  const float* bq = (const float*)d_in[9];
  const float* Wo = (const float*)d_in[10]; const float* bo = (const float*)d_in[11];
  float* out = (float*)d_out;

  const int M = BATCH * SEQ;                    // 8192
  const size_t actE = (size_t)M * EMBED;        // 8M elements
  const size_t wE   = (size_t)EMBED * EMBED;    // 1M elements

  bf16_t* qx   = (bf16_t*)d_ws;      // converted activations (dead after projections)
  bf16_t* kx   = qx + actE;
  bf16_t* vx   = kx + actE;
  bf16_t* wqb  = vx + actE;          // converted weights
  bf16_t* wkb  = wqb + wE;
  bf16_t* wvb  = wkb + wE;
  bf16_t* wob  = wvb + wE;
  bf16_t* qbuf = wob + wE;           // projected q/k/v
  bf16_t* kbuf = qbuf + actE;
  bf16_t* vbuf = kbuf + actE;
  bf16_t* abuf = qx;                 // attention out aliases dead qx region

  // f32 -> bf16 conversion passes
  cvt_f32_to_bf16<<<(int)(actE / 4 / 256), 256, 0, stream>>>(query,  qx, (int)(actE / 4));
  cvt_f32_to_bf16<<<(int)(actE / 4 / 256), 256, 0, stream>>>(keys,   kx, (int)(actE / 4));
  cvt_f32_to_bf16<<<(int)(actE / 4 / 256), 256, 0, stream>>>(values, vx, (int)(actE / 4));
  cvt_f32_to_bf16<<<(int)(wE / 4 / 256), 256, 0, stream>>>(Wq, wqb, (int)(wE / 4));
  cvt_f32_to_bf16<<<(int)(wE / 4 / 256), 256, 0, stream>>>(Wk, wkb, (int)(wE / 4));
  cvt_f32_to_bf16<<<(int)(wE / 4 / 256), 256, 0, stream>>>(Wv, wvb, (int)(wE / 4));
  cvt_f32_to_bf16<<<(int)(wE / 4 / 256), 256, 0, stream>>>(Wo, wob, (int)(wE / 4));

  dim3 gridG(EMBED / 128, M / 128);
  gemm_bias_wmma<true><<<gridG, 256, 0, stream>>>(qx, wqb, bq, qbuf, M, EMBED, EMBED);
  gemm_bias_wmma<true><<<gridG, 256, 0, stream>>>(kx, wkb, bk, kbuf, M, EMBED, EMBED);
  gemm_bias_wmma<true><<<gridG, 256, 0, stream>>>(vx, wvb, bv, vbuf, M, EMBED, EMBED);

  dim3 gridA(SEQ / 128, HEADS, BATCH);
  flash_attn_wmma<<<gridA, 256, 0, stream>>>(qbuf, kbuf, vbuf, mask, abuf);

  gemm_bias_wmma<false><<<gridG, 256, 0, stream>>>(abuf, wob, bo, out, M, EMBED, EMBED);
}
